// ContextualLoss_37065567765250
// MI455X (gfx1250) — matrix operand order repaired and checked by reading, not compiled, not run on hardware
//
#include <hip/hip_runtime.h>
#include <hip/hip_bf16.h>

typedef __attribute__((ext_vector_type(16))) _Float16 v16h;
typedef __attribute__((ext_vector_type(8)))  _Float16 v8h;
typedef __attribute__((ext_vector_type(8)))  float    v8f;

#define CCH     256          // channels (K of the GEMM)
#define HW      9216         // 96*96 spatial positions (M and N)
#define NBLK32  (HW / 32)    // 288 32-wide register-blocked tiles
#define CSPLIT  4            // waves splitting the streamed dimension
#define SEG     (HW / CSPLIT)     // 2304
#define SEG_TILES (SEG / 16)      // 144
#define EPS_REL  1e-5f
#define EPS_NORM 1e-12f

// ---------------- fragment loaders (ISA 7.12.2 layouts) ----------------
// A (16x32 f16, MxK): lanes 0-15 row M=lane, K = k0+{0..7,16..23};
//                     lanes 16-31 row M=lane-16, K = k0+{8..15,24..31}.
__device__ inline v16h load_a_frag(const _Float16* __restrict__ base,
                                   int row0, int k0, int lane) {
  int m  = lane & 15;
  int hi = lane >> 4;
  const _Float16* p = base + (size_t)(row0 + m) * CCH + k0 + hi * 8;
  v8h lo = *(const v8h*)(p);
  v8h hh = *(const v8h*)(p + 16);
  v16h r;
#pragma unroll
  for (int i = 0; i < 8; ++i) { r[i] = lo[i]; r[i + 8] = hh[i]; }
  return r;
}

// B (32x16 f16, KxN): lanes 0-15 col N=lane, K = k0+{0..7,16..23};
// lanes 16-31 col N=lane-16, K = k0+{8..15,24..31} (matches A).
// Bfeat is stored as [col][k] row-major, so this is two 16B runs.
__device__ inline v16h load_b_frag(const _Float16* __restrict__ base,
                                   int col0, int k0, int lane) {
  int n  = lane & 15;
  int hi = lane >> 4;
  const _Float16* p = base + (size_t)(col0 + n) * CCH + k0 + hi * 8;
  v8h lo = *(const v8h*)(p);
  v8h hh = *(const v8h*)(p + 16);
  v16h r;
#pragma unroll
  for (int i = 0; i < 8; ++i) { r[i] = lo[i]; r[i + 8] = hh[i]; }
  return r;
}

// ---------------- stage 1: per-channel mean of gt ----------------
__global__ void cx_mean(const float* __restrict__ gt, float* __restrict__ meanT) {
  __shared__ float s[256];
  int c = blockIdx.x;
  float acc = 0.0f;
  for (int i = threadIdx.x; i < HW; i += 256)
    acc += gt[(size_t)c * HW + i];
  s[threadIdx.x] = acc;
  __syncthreads();
  for (int st = 128; st > 0; st >>= 1) {
    if (threadIdx.x < st) s[threadIdx.x] += s[threadIdx.x + st];
    __syncthreads();
  }
  if (threadIdx.x == 0) meanT[c] = s[0] / (float)HW;
}

// ---------------- stage 2: center, L2-normalize, pack f16 [pos][chan] ----
__global__ void cx_normalize(const float* __restrict__ img,
                             const float* __restrict__ gt,
                             const float* __restrict__ meanT,
                             _Float16* __restrict__ Af,
                             _Float16* __restrict__ Bf) {
  __shared__ float sm[CCH];
  sm[threadIdx.x] = meanT[threadIdx.x];   // blockDim == 256 == CCH
  __syncthreads();
  int p = blockIdx.x * 256 + threadIdx.x; // grid = 36 -> covers 9216 positions
  float si = 0.0f, st = 0.0f;
  for (int c = 0; c < CCH; ++c) {
    float m  = sm[c];
    float xi = img[(size_t)c * HW + p] - m; si += xi * xi;
    float xt = gt [(size_t)c * HW + p] - m; st += xt * xt;
  }
  float ii = 1.0f / fmaxf(sqrtf(si), EPS_NORM);
  float it = 1.0f / fmaxf(sqrtf(st), EPS_NORM);
  for (int c = 0; c < CCH; c += 8) {
    v8h a, b;
#pragma unroll
    for (int j = 0; j < 8; ++j) {
      a[j] = (_Float16)((img[(size_t)(c + j) * HW + p] - sm[c + j]) * ii);
      b[j] = (_Float16)((gt [(size_t)(c + j) * HW + p] - sm[c + j]) * it);
    }
    *(v8h*)(Af + (size_t)p * CCH + c) = a;
    *(v8h*)(Bf + (size_t)p * CCH + c) = b;
  }
}

// ---------------- stage 3: init side arrays ----------------
__global__ void cx_init(float* __restrict__ rowmin, float* __restrict__ rowsum,
                        float* __restrict__ colmax) {
  int i = blockIdx.x * 256 + threadIdx.x;
  if (i < HW) {
    ((unsigned*)rowmin)[i] = 0x7F800000u; // +inf
    rowsum[i] = 0.0f;
    colmax[i] = 0.0f;
  }
}

// -------- in-place per-row reciprocal prep (hoists divisions out of GEMMs)
// MODE 0: rowmin[r] <- 2/(rowmin[r]+eps)   (run after pass_rows<0>)
// MODE 1: rowsum[r] <- 1/rowsum[r]         (run after pass_rows<1>)
template <int MODE>
__global__ void cx_recip(float* __restrict__ x) {
  int i = blockIdx.x * 256 + threadIdx.x;
  if (i < HW) {
    if (MODE == 0) x[i] = 2.0f / (x[i] + EPS_REL);
    else           x[i] = 1.0f / x[i];
  }
}

// ---------------- stages 4/5: WMMA GEMM (2x1 register-blocked) -----------
// MODE 0 = per-row min of dist; MODE 1 = per-row sum of exp(2 - d*invm).
// Each wave keeps TWO 16-row A blocks (16 fragments) resident and streams B,
// so every streamed B fragment feeds two WMMAs (halves cache traffic/WMMA).
template <int MODE>
__global__ __launch_bounds__(128) void cx_pass_rows(
    const _Float16* __restrict__ A, const _Float16* __restrict__ B,
    float* __restrict__ rowmin, float* __restrict__ rowsum) {
  int wave   = blockIdx.x * 4 + (threadIdx.x >> 5);
  int lane   = threadIdx.x & 31;
  int hi     = lane >> 4;
  int supRow = wave / CSPLIT;          // 0..287
  int colSeg = wave % CSPLIT;
  int row0   = supRow * 32;            // first block rows row0..row0+15
  int row1   = row0 + 16;              // second block rows row1..row1+15

  v16h afrag0[8], afrag1[8];
#pragma unroll
  for (int k = 0; k < 8; ++k) {
    afrag0[k] = load_a_frag(A, row0, k * 32, lane);
    afrag1[k] = load_a_frag(A, row1, k * 32, lane);
  }

  float invm0[8], invm1[8];
  if (MODE == 1) {    // rowmin[] already holds 2/(min+eps) after cx_recip<0>
#pragma unroll
    for (int v = 0; v < 8; ++v) {
      invm0[v] = rowmin[row0 + v + 8 * hi];
      invm1[v] = rowmin[row1 + v + 8 * hi];
    }
  }
  float racc0[8], racc1[8];
#pragma unroll
  for (int v = 0; v < 8; ++v) {
    racc0[v] = (MODE == 0) ? 3.0e38f : 0.0f;
    racc1[v] = (MODE == 0) ? 3.0e38f : 0.0f;
  }

  for (int t = 0; t < SEG_TILES; ++t) {
    int col0 = colSeg * SEG + t * 16;
    v8f acc0 = {}, acc1 = {};
#pragma unroll
    for (int k = 0; k < 8; ++k) {
      v16h b = load_b_frag(B, col0, k * 32, lane);
      acc0 = __builtin_amdgcn_wmma_f32_16x16x32_f16(
          false, afrag0[k], false, b, (short)0, acc0, false, false);
      acc1 = __builtin_amdgcn_wmma_f32_16x16x32_f16(
          false, afrag1[k], false, b, (short)0, acc1, false, false);
    }
#pragma unroll
    for (int v = 0; v < 8; ++v) {
      float d0 = fmaxf(0.0f, (1.0f - acc0[v]) * 0.5f);
      float d1 = fmaxf(0.0f, (1.0f - acc1[v]) * 0.5f);
      if (MODE == 0) {
        racc0[v] = fminf(racc0[v], d0);
        racc1[v] = fminf(racc1[v], d1);
      } else {
        racc0[v] += __expf(2.0f - d0 * invm0[v]);
        racc1[v] += __expf(2.0f - d1 * invm1[v]);
      }
    }
  }
  // combine the 16 columns held across lanes of each half-wave
#pragma unroll
  for (int v = 0; v < 8; ++v) {
#pragma unroll
    for (int m = 1; m < 16; m <<= 1) {
      float o0 = __shfl_xor(racc0[v], m, 32);
      float o1 = __shfl_xor(racc1[v], m, 32);
      racc0[v] = (MODE == 0) ? fminf(racc0[v], o0) : (racc0[v] + o0);
      racc1[v] = (MODE == 0) ? fminf(racc1[v], o1) : (racc1[v] + o1);
    }
  }
  if ((lane & 15) == 0) {
#pragma unroll
    for (int v = 0; v < 8; ++v) {
      int r0 = row0 + v + 8 * hi;
      int r1 = row1 + v + 8 * hi;
      if (MODE == 0) {
        atomicMin((unsigned*)&rowmin[r0], __float_as_uint(racc0[v]));
        atomicMin((unsigned*)&rowmin[r1], __float_as_uint(racc1[v]));
      } else {
        atomicAdd(&rowsum[r0], racc0[v]);
        atomicAdd(&rowsum[r1], racc1[v]);
      }
    }
  }
}

// ---------------- stage 6: WMMA GEMM, per-column max of cx (1x2 blocked) --
// invm[] = 2/(rowmin+eps), invs[] = 1/rowsum (precomputed by cx_recip).
__global__ __launch_bounds__(128) void cx_pass_cols(
    const _Float16* __restrict__ A, const _Float16* __restrict__ B,
    const float* __restrict__ invm, const float* __restrict__ invs,
    float* __restrict__ colmax) {
  int wave   = blockIdx.x * 4 + (threadIdx.x >> 5);
  int lane   = threadIdx.x & 31;
  int hi     = lane >> 4;
  int supCol = wave / CSPLIT;          // 0..287
  int rowSeg = wave % CSPLIT;
  int col0   = supCol * 32;
  int col1   = col0 + 16;

  v16h bfrag0[8], bfrag1[8];
#pragma unroll
  for (int k = 0; k < 8; ++k) {
    bfrag0[k] = load_b_frag(B, col0, k * 32, lane);
    bfrag1[k] = load_b_frag(B, col1, k * 32, lane);
  }

  float cmax0 = 0.0f, cmax1 = 0.0f;
  for (int t = 0; t < SEG_TILES; ++t) {
    int row0 = rowSeg * SEG + t * 16;
    v8f acc0 = {}, acc1 = {};
#pragma unroll
    for (int k = 0; k < 8; ++k) {
      v16h a = load_a_frag(A, row0, k * 32, lane);
      acc0 = __builtin_amdgcn_wmma_f32_16x16x32_f16(
          false, a, false, bfrag0[k], (short)0, acc0, false, false);
      acc1 = __builtin_amdgcn_wmma_f32_16x16x32_f16(
          false, a, false, bfrag1[k], (short)0, acc1, false, false);
    }
#pragma unroll
    for (int v = 0; v < 8; ++v) {
      int r = row0 + v + 8 * hi;
      float im = invm[r];
      float is = invs[r];
      float d0 = fmaxf(0.0f, (1.0f - acc0[v]) * 0.5f);
      float d1 = fmaxf(0.0f, (1.0f - acc1[v]) * 0.5f);
      cmax0 = fmaxf(cmax0, __expf(2.0f - d0 * im) * is);
      cmax1 = fmaxf(cmax1, __expf(2.0f - d1 * im) * is);
    }
  }
  cmax0 = fmaxf(cmax0, __shfl_xor(cmax0, 16, 32));
  cmax1 = fmaxf(cmax1, __shfl_xor(cmax1, 16, 32));
  if (lane < 16) {
    atomicMax((unsigned*)&colmax[col0 + lane], __float_as_uint(cmax0));
    atomicMax((unsigned*)&colmax[col1 + lane], __float_as_uint(cmax1));
  }
}

// ---------------- stage 7: loss = -log(mean(colmax)) ----------------
__global__ void cx_finalize(const float* __restrict__ colmax,
                            float* __restrict__ out) {
  __shared__ float s[256];
  float acc = 0.0f;
  for (int i = threadIdx.x; i < HW; i += 256) acc += colmax[i];
  s[threadIdx.x] = acc;
  __syncthreads();
  for (int st = 128; st > 0; st >>= 1) {
    if (threadIdx.x < st) s[threadIdx.x] += s[threadIdx.x + st];
    __syncthreads();
  }
  if (threadIdx.x == 0) {
    float cs = s[0] / (float)HW;
    out[0] = -logf(cs);
  }
}

// ---------------- host launcher ----------------
extern "C" void kernel_launch(void* const* d_in, const int* in_sizes, int n_in,
                              void* d_out, int out_size, void* d_ws, size_t ws_size,
                              hipStream_t stream) {
  (void)in_sizes; (void)n_in; (void)out_size; (void)ws_size;
  const float* img = (const float*)d_in[0];
  const float* gt  = (const float*)d_in[1];
  float* out = (float*)d_out;

  char* w = (char*)d_ws;
  float* meanT  = (float*)w; w += 1024;                 // 256 f32 (padded)
  float* rowmin = (float*)w; w += (size_t)HW * 4;       // becomes invm
  float* rowsum = (float*)w; w += (size_t)HW * 4;       // becomes invs
  float* colmax = (float*)w; w += (size_t)HW * 4;
  _Float16* Af = (_Float16*)w; w += (size_t)HW * CCH * 2;
  _Float16* Bf = (_Float16*)w; w += (size_t)HW * CCH * 2;

  cx_mean<<<CCH, 256, 0, stream>>>(gt, meanT);
  cx_normalize<<<HW / 256, 256, 0, stream>>>(img, gt, meanT, Af, Bf);
  cx_init<<<HW / 256, 256, 0, stream>>>(rowmin, rowsum, colmax);
  cx_pass_rows<0><<<NBLK32, 128, 0, stream>>>(Af, Bf, rowmin, rowsum);
  cx_recip<0><<<HW / 256, 256, 0, stream>>>(rowmin);          // rowmin -> invm
  cx_pass_rows<1><<<NBLK32, 128, 0, stream>>>(Af, Bf, rowmin, rowsum);
  cx_recip<1><<<HW / 256, 256, 0, stream>>>(rowsum);          // rowsum -> invs
  cx_pass_cols<<<NBLK32, 128, 0, stream>>>(Af, Bf, rowmin, rowsum, colmax);
  cx_finalize<<<1, 256, 0, stream>>>(colmax, out);
}
